// GraphormerMultiHeadAttention_23106924052832
// MI455X (gfx1250) — matrix operand order repaired and checked by reading, not compiled, not run on hardware
//
#include <hip/hip_runtime.h>
#include <cstdint>
#include <cstddef>

// ---------------------------------------------------------------------------
// Graphormer multi-head attention for gfx1250 (MI455X), wave32 + WMMA bf16.
// N=3072 nodes, D=256, H=8 heads (hd=32), E=98304 edges, B=16 graphs.
// ---------------------------------------------------------------------------

constexpr int kN = 3072;   // nodes
constexpr int kD = 256;    // embed dim
constexpr int kH = 8;      // heads
constexpr int kHD = 32;    // head dim
constexpr int kE = 98304;  // edges
constexpr int kKSplit = 8; // K-split for attn@V

typedef __attribute__((ext_vector_type(16))) __bf16 v16bf;
typedef __attribute__((ext_vector_type(8)))  __bf16 v8bf;
typedef __attribute__((ext_vector_type(8)))  float  v8f;

union BF16x16 { v16bf v; v8bf h[2]; };

// ---- WMMA fragment loaders (layouts per CDNA5 ISA 7.12.2, wave32) ----------
// A 16x32 bf16 (MxK), row-major source, stride lda (elements):
//   lane L: m = L%16, hi = L/16 ; halfs 0..7  -> K = 8*hi + t
//                                 halfs 8..15 -> K = 16 + 8*hi + (t-8)
__device__ inline v16bf load_a_bf16(const __bf16* __restrict__ base, int lda) {
  const int lane = threadIdx.x & 31;
  const int m = lane & 15, hi = lane >> 4;
  const __bf16* p = base + (size_t)m * lda + 8 * hi;
  BF16x16 r;
  r.h[0] = *(const v8bf*)(p);
  r.h[1] = *(const v8bf*)(p + 16);
  return r.v;
}

// Same A layout but converting from f32 source on the fly.
__device__ inline v16bf load_a_f32(const float* __restrict__ base, int lda) {
  const int lane = threadIdx.x & 31;
  const int m = lane & 15, hi = lane >> 4;
  const float* p = base + (size_t)m * lda + 8 * hi;
  BF16x16 r;
#pragma unroll
  for (int t = 0; t < 8; ++t) r.v[t] = (__bf16)p[t];
#pragma unroll
  for (int t = 0; t < 8; ++t) r.v[8 + t] = (__bf16)p[16 + t];
  return r.v;
}

// B 32x16 bf16 (KxN) supplied in K-contiguous ("B^T") layout bt[n][k],
// stride ldb: lane L: n = L%16, hk = L/16 ; halfs 0..7 -> K = 8*hk + t, halfs 8..15 -> K = 16 + 8*hk + (t-8).
__device__ inline v16bf load_bT_bf16(const __bf16* __restrict__ base, int ldb) {
  const int lane = threadIdx.x & 31;
  const int n = lane & 15, hk = lane >> 4;
  const __bf16* p = base + (size_t)n * ldb + 8 * hk;
  BF16x16 r;
  r.h[0] = *(const v8bf*)(p);
  r.h[1] = *(const v8bf*)(p + 16);
  return r.v;
}

__device__ inline v8f wmma_bf16(v16bf a, v16bf b, v8f c) {
  // (neg_a, A, neg_b, B, c_mod, C, reuse_a, reuse_b)
  return __builtin_amdgcn_wmma_f32_16x16x32_bf16(
      false, a, false, b, (short)0, c, false, false);
}

// ---------------------------------------------------------------------------
// Elementwise prep kernels
// ---------------------------------------------------------------------------
__global__ void __launch_bounds__(256) cvt_bf16_kernel(
    const float* __restrict__ in, __bf16* __restrict__ out, int n) {
  int i = blockIdx.x * 256 + threadIdx.x;
  if (i < n) out[i] = (__bf16)in[i];
}

// wT[n*d + k] = (bf16) w[k*d + n]   (D x D)
__global__ void __launch_bounds__(256) transpose_bf16_kernel(
    const float* __restrict__ w, __bf16* __restrict__ wT, int d) {
  int i = blockIdx.x * 256 + threadIdx.x;
  if (i < d * d) {
    int k = i / d, n = i % d;
    wT[(size_t)n * d + k] = (__bf16)w[i];
  }
}

__global__ void __launch_bounds__(256) zero_f32_kernel(
    float* __restrict__ p, int n) {
  int i = blockIdx.x * 256 + threadIdx.x;
  if (i < n) p[i] = 0.0f;
}

// eb[e*H + h] = edge_attr[e,:4] @ We[:, h] + be[h]
__global__ void __launch_bounds__(256) edge_bias_kernel(
    const float* __restrict__ ea, const float* __restrict__ We,
    const float* __restrict__ be, float* __restrict__ eb) {
  int i = blockIdx.x * 256 + threadIdx.x;
  if (i >= kE * kH) return;
  int e = i / kH, h = i % kH;
  float s = be[h];
#pragma unroll
  for (int j = 0; j < 4; ++j) s += ea[e * 4 + j] * We[j * kH + h];
  eb[i] = s;
}

// ---------------------------------------------------------------------------
// Projection GEMM: out = xb @ W + b (16x16 tile per wave, K-loop of WMMAs).
// vt_mode==0: out is bf16 [N,D] row-major (Q, K).
// vt_mode==1: out is bf16 [D,N] transposed (V^T: K-contiguous for attn@V B).
// ---------------------------------------------------------------------------
__global__ void __launch_bounds__(32) proj_gemm_kernel(
    const __bf16* __restrict__ xb, const __bf16* __restrict__ wT,
    const float* __restrict__ bias, __bf16* __restrict__ out, int vt_mode) {
  const int lane = threadIdx.x & 31;
  const int hi = lane >> 4;
  const int m0 = blockIdx.x * 16, n0 = blockIdx.y * 16;
  v8f c = {};
  for (int k0 = 0; k0 < kD; k0 += 32) {
    v16bf a = load_a_bf16(xb + (size_t)m0 * kD + k0, kD);
    v16bf b = load_bT_bf16(wT + (size_t)n0 * kD + k0, kD);
    c = wmma_bf16(a, b, c);
  }
  const int col = n0 + (lane & 15);
  const float bb = bias[col];
#pragma unroll
  for (int r = 0; r < 8; ++r) {
    int m = m0 + r + 8 * hi;
    float val = c[r] + bb;
    if (vt_mode) out[(size_t)col * kN + m] = (__bf16)val;
    else         out[(size_t)m * kD + col] = (__bf16)val;
  }
}

// ---------------------------------------------------------------------------
// Per-head scores: sc[m,n] = (Q_h[m,:] . K_h[n,:]) / sqrt(32), masked to -inf
// across graphs. One v_wmma per 16x16 tile (K = hd = 32).
// ---------------------------------------------------------------------------
__global__ void __launch_bounds__(32) scores_kernel(
    const __bf16* __restrict__ Qb, const __bf16* __restrict__ Kb,
    const int* __restrict__ batch, float* __restrict__ sc, int head) {
  const int lane = threadIdx.x & 31;
  const int hi = lane >> 4;
  const int m0 = blockIdx.x * 16, n0 = blockIdx.y * 16;
  v16bf a = load_a_bf16(Qb + (size_t)m0 * kD + head * kHD, kD);
  v16bf b = load_bT_bf16(Kb + (size_t)n0 * kD + head * kHD, kD);
  v8f c = {};
  c = wmma_bf16(a, b, c);
  const int col = n0 + (lane & 15);
  const int bc = batch[col];
  const float scale = 0.17677669529663687f;  // 1/sqrt(32)
#pragma unroll
  for (int r = 0; r < 8; ++r) {
    int m = m0 + r + 8 * hi;
    float s = (batch[m] == bc) ? c[r] * scale : -__builtin_huge_valf();
    sc[(size_t)m * kN + col] = s;
  }
}

// Scatter-add edge bias into this head's scores (duplicates accumulate).
__global__ void __launch_bounds__(256) scatter_bias_kernel(
    const int* __restrict__ ei, const float* __restrict__ eb,
    float* __restrict__ sc, int head) {
  int e = blockIdx.x * 256 + threadIdx.x;
  if (e >= kE) return;
  int s = ei[e], d = ei[kE + e];
  atomicAdd(&sc[(size_t)s * kN + d], eb[e * kH + head]);
}

// Row softmax: f32 scores -> bf16 attention probabilities.
__global__ void __launch_bounds__(256) softmax_kernel(
    const float* __restrict__ sc, __bf16* __restrict__ at) {
  __shared__ float red[256];
  const int row = blockIdx.x;
  const float* s = sc + (size_t)row * kN;
  float vals[kN / 256];
  float m = -__builtin_huge_valf();
#pragma unroll
  for (int j = 0; j < kN / 256; ++j) {
    vals[j] = s[threadIdx.x + j * 256];
    m = fmaxf(m, vals[j]);
  }
  red[threadIdx.x] = m; __syncthreads();
  for (int o = 128; o > 0; o >>= 1) {
    if (threadIdx.x < o) red[threadIdx.x] = fmaxf(red[threadIdx.x], red[threadIdx.x + o]);
    __syncthreads();
  }
  m = red[0]; __syncthreads();
  float sum = 0.0f;
#pragma unroll
  for (int j = 0; j < kN / 256; ++j) {
    vals[j] = __expf(vals[j] - m);
    sum += vals[j];
  }
  red[threadIdx.x] = sum; __syncthreads();
  for (int o = 128; o > 0; o >>= 1) {
    if (threadIdx.x < o) red[threadIdx.x] += red[threadIdx.x + o];
    __syncthreads();
  }
  const float inv = 1.0f / red[0];
  __bf16* arow = at + (size_t)row * kN;
#pragma unroll
  for (int j = 0; j < kN / 256; ++j)
    arow[threadIdx.x + j * 256] = (__bf16)(vals[j] * inv);
}

// ---------------------------------------------------------------------------
// attn @ V_h with K-split over the 3072 column-node dimension; partial f32
// accumulated into attn_out[N,D] with global atomics (zeroed each call).
// ---------------------------------------------------------------------------
__global__ void __launch_bounds__(32) av_kernel(
    const __bf16* __restrict__ at, const __bf16* __restrict__ Vt,
    float* __restrict__ ao, int head) {
  const int lane = threadIdx.x & 31;
  const int hi = lane >> 4;
  const int m0 = blockIdx.x * 16;
  const int d0 = blockIdx.y * 16;
  const int kb = blockIdx.z * (kN / kKSplit);
  const int ke = kb + (kN / kKSplit);
  const __bf16* vbase = Vt + (size_t)(head * kHD + d0) * kN;
  const __bf16* abase = at + (size_t)m0 * kN;
  v8f c = {};
  for (int k0 = kb; k0 < ke; k0 += 32) {
    if (k0 + 32 < ke) {
      __builtin_prefetch(abase + (size_t)(lane & 15) * kN + k0 + 32, 0, 1);
      __builtin_prefetch(vbase + (size_t)(lane & 15) * kN + k0 + 32, 0, 1);
    }
    v16bf a = load_a_bf16(abase + k0, kN);
    v16bf b = load_bT_bf16(vbase + k0, kN);
    c = wmma_bf16(a, b, c);
  }
  const int col = head * kHD + d0 + (lane & 15);
#pragma unroll
  for (int r = 0; r < 8; ++r) {
    int m = m0 + r + 8 * hi;
    atomicAdd(&ao[(size_t)m * kD + col], c[r]);
  }
}

// Final projection: y = attn_out @ Wo + bo (f32 in, bf16 WMMA, f32 out).
__global__ void __launch_bounds__(32) out_gemm_kernel(
    const float* __restrict__ ao, const __bf16* __restrict__ woT,
    const float* __restrict__ bo, float* __restrict__ y) {
  const int lane = threadIdx.x & 31;
  const int hi = lane >> 4;
  const int m0 = blockIdx.x * 16, n0 = blockIdx.y * 16;
  v8f c = {};
  for (int k0 = 0; k0 < kD; k0 += 32) {
    v16bf a = load_a_f32(ao + (size_t)m0 * kD + k0, kD);
    v16bf b = load_bT_bf16(woT + (size_t)n0 * kD + k0, kD);
    c = wmma_bf16(a, b, c);
  }
  const int col = n0 + (lane & 15);
  const float bb = bo[col];
#pragma unroll
  for (int r = 0; r < 8; ++r) {
    int m = m0 + r + 8 * hi;
    y[(size_t)m * kD + col] = c[r] + bb;
  }
}

// ---------------------------------------------------------------------------
// Host launcher
// ---------------------------------------------------------------------------
extern "C" void kernel_launch(void* const* d_in, const int* in_sizes, int n_in,
                              void* d_out, int out_size, void* d_ws, size_t ws_size,
                              hipStream_t stream) {
  (void)in_sizes; (void)n_in; (void)out_size; (void)ws_size;
  const float* x   = (const float*)d_in[0];
  const float* ea  = (const float*)d_in[1];
  const float* Wq  = (const float*)d_in[2];  const float* bq = (const float*)d_in[3];
  const float* Wk  = (const float*)d_in[4];  const float* bk = (const float*)d_in[5];
  const float* Wv  = (const float*)d_in[6];  const float* bv = (const float*)d_in[7];
  const float* Wo  = (const float*)d_in[8];  const float* bo = (const float*)d_in[9];
  const float* We  = (const float*)d_in[10]; const float* be = (const float*)d_in[11];
  const int* ei    = (const int*)d_in[12];
  const int* batch = (const int*)d_in[13];
  float* y = (float*)d_out;

  // Workspace carve (256B aligned); ~70 MB total.
  char* base = (char*)d_ws;
  size_t off = 0;
  auto carve = [&](size_t bytes) -> void* {
    void* r = base + off;
    off += (bytes + 255) & ~(size_t)255;
    return r;
  };
  __bf16* xb  = (__bf16*)carve((size_t)kN * kD * 2);
  __bf16* wqT = (__bf16*)carve((size_t)kD * kD * 2);
  __bf16* wkT = (__bf16*)carve((size_t)kD * kD * 2);
  __bf16* wvT = (__bf16*)carve((size_t)kD * kD * 2);
  __bf16* woT = (__bf16*)carve((size_t)kD * kD * 2);
  __bf16* Qb  = (__bf16*)carve((size_t)kN * kD * 2);
  __bf16* Kb  = (__bf16*)carve((size_t)kN * kD * 2);
  __bf16* Vt  = (__bf16*)carve((size_t)kD * kN * 2);   // [D][N] = per-head V^T
  float*  eb  = (float*) carve((size_t)kE * kH * 4);
  float*  ao  = (float*) carve((size_t)kN * kD * 4);
  float*  sc  = (float*) carve((size_t)kN * kN * 4);   // per-head scores
  __bf16* at  = (__bf16*)carve((size_t)kN * kN * 2);   // per-head attn (bf16)

  // Prep
  cvt_bf16_kernel<<<(kN * kD) / 256, 256, 0, stream>>>(x, xb, kN * kD);
  transpose_bf16_kernel<<<(kD * kD) / 256, 256, 0, stream>>>(Wq, wqT, kD);
  transpose_bf16_kernel<<<(kD * kD) / 256, 256, 0, stream>>>(Wk, wkT, kD);
  transpose_bf16_kernel<<<(kD * kD) / 256, 256, 0, stream>>>(Wv, wvT, kD);
  transpose_bf16_kernel<<<(kD * kD) / 256, 256, 0, stream>>>(Wo, woT, kD);
  zero_f32_kernel<<<(kN * kD) / 256, 256, 0, stream>>>(ao, kN * kD);
  edge_bias_kernel<<<(kE * kH) / 256, 256, 0, stream>>>(ea, We, be, eb);

  // Q/K/V projections (WMMA)
  dim3 gproj(kN / 16, kD / 16);
  proj_gemm_kernel<<<gproj, 32, 0, stream>>>(xb, wqT, bq, Qb, 0);
  proj_gemm_kernel<<<gproj, 32, 0, stream>>>(xb, wkT, bk, Kb, 0);
  proj_gemm_kernel<<<gproj, 32, 0, stream>>>(xb, wvT, bv, Vt, 1);

  // Per-head attention (stream order provides all dependencies)
  dim3 gsc(kN / 16, kN / 16);
  dim3 gav(kN / 16, kHD / 16, kKSplit);
  for (int h = 0; h < kH; ++h) {
    scores_kernel<<<gsc, 32, 0, stream>>>(Qb, Kb, batch, sc, h);
    scatter_bias_kernel<<<kE / 256, 256, 0, stream>>>(ei, eb, sc, h);
    softmax_kernel<<<kN, 256, 0, stream>>>(sc, at);
    av_kernel<<<gav, 32, 0, stream>>>(at, Vt, ao, h);
  }

  // Output projection
  out_gemm_kernel<<<gproj, 32, 0, stream>>>(ao, woT, bo, y);
}
